// Snake_TCN_68521908240994
// MI455X (gfx1250) — matrix operand order repaired and checked by reading, not compiled, not run on hardware
//
#include <hip/hip_runtime.h>
#include <hip/hip_bf16.h>
#include <math.h>

// ---------------------------------------------------------------------------
// Snake-TCN fused kernel for gfx1250 (MI455X).
// DFT / proj / MLP all lowered to v_wmma_f32_16x16x32_f16.
// Weights pre-packed to f16 once per launch; B fragments are single 32B loads.
// Fast TRANS-unit math (v_sqrt/v_log/v_rcp/v_exp/v_sin) for elementwise ops.
// ---------------------------------------------------------------------------

typedef __attribute__((ext_vector_type(16))) _Float16 v16h;
typedef __attribute__((ext_vector_type(8)))  _Float16 v8h;
typedef __attribute__((ext_vector_type(8)))  float    v8f;

#define NB_    64      // batch
#define NC_    8       // channels
#define NT_    2048    // time
#define NK_    64      // kernel / FFT length
#define NF_    32      // freq bins kept
#define MLP_   32
#define HID_   64
#define FD_    512     // FFT_DIM = C*NF*2
#define NW_    1985    // W = T - (K-1)
#define NTILES_ ((NW_ + 15) / 16)   // 125
#define WINLEN_ 80     // staged window (79 used, padded to 80 for float4 loads)

// d_ws layout (bytes): [0) wts f32[2048] | 8192) pwh f16[32*512]
//                      | 40960) w0h f16[64*32] | 45056) w1h f16[32*64] | 49152 total
#define WS_PWH_OFF_  8192
#define WS_W0H_OFF_  (WS_PWH_OFF_ + MLP_ * FD_ * 2)
#define WS_W1H_OFF_  (WS_W0H_OFF_ + HID_ * MLP_ * 2)

// --- WMMA f16 16x16x32 fragment index maps (per CDNA5 ISA 7.12.2) ----------
// A (16x32, 16-bit): lanes 0-15 hold K=0..7 (v0-3) and K=16..23 (v4-7);
//                    lanes 16-31 hold K=8..15 and K=24..31. Row M = lane%16.
__device__ __forceinline__ int amapK(int i, int hi) {
  return (i < 8) ? ((hi ? 8 : 0) + i) : ((hi ? 24 : 16) + (i - 8));
}
// B (32x16, 16-bit): col N = lane%16; this code feeds K contiguously (A too): lanes 0-15 K=0..15, 16-31 K=16..31.
// => B-fragment K range is 16 contiguous values starting at hi*16.
// C/D (16x16 f32): col N = lane%16, row M = vgpr + 8*(lane>=16).

__device__ __forceinline__ v8f wmma16(v16h a, v16h b, v8f c) {
  return __builtin_amdgcn_wmma_f32_16x16x32_f16(
      /*neg_a=*/false, a, /*neg_b=*/false, b,
      /*c_mod=*/(short)0, c, /*reuse_a=*/false, /*reuse_b=*/false);
}

// B fragment = 16 contiguous pre-packed f16 weights (32B-aligned) -> one
// 32-byte global load (2x global_load_b128), no conversion ALU.
__device__ __forceinline__ v16h ldgBh(const _Float16* __restrict__ p) {
  return *(const v16h*)p;
}

// Load an A fragment from a 16B-aligned LDS row (rowk = row base + 32*kchunk):
// lane half hi holds K = 16*hi + s, two 8-half groups at 16B offsets 16*hi and 16*hi+8.
__device__ __forceinline__ v16h ldsA(const _Float16* rowk, int hi) {
  const v8h* p = (const v8h*)rowk;
  const v8h lo = p[hi ? 2 : 0];   // K = 0..7  or 16..23
  const v8h h8 = p[hi ? 3 : 1];   // K = 8..15 or 24..31
  v16h r;
#pragma unroll
  for (int q = 0; q < 8; ++q) { r[q] = lo[q]; r[8 + q] = h8[q]; }
  return r;
}

// --- fast elementwise math on TRANS units ----------------------------------
__device__ __forceinline__ float fast_rcp(float x)  { return __builtin_amdgcn_rcpf(x); }
__device__ __forceinline__ float fast_sqrt(float x) { return __builtin_amdgcn_sqrtf(x); }

// atan2(y,x)/pi via 4-term minimax atan poly + quadrant fixups in pi units.
__device__ __forceinline__ float atan2_over_pi(float y, float x) {
  const float ax = fabsf(x), ay = fabsf(y);
  const float mx = fmaxf(ax, ay), mn = fminf(ax, ay);
  const float a = mn * fast_rcp(fmaxf(mx, 1e-30f));
  const float s = a * a;
  float r = ((-0.0464964749f * s + 0.15931422f) * s - 0.327622764f) * s * a + a;
  r *= 0.31830988618379067154f;            // -> pi units
  if (ay > ax)   r = 0.5f - r;
  if (x < 0.0f)  r = 1.0f - r;
  return (y < 0.0f) ? -r : r;
}

__device__ __forceinline__ float fast_tanh(float x) {
  x = fminf(fmaxf(x, -15.0f), 15.0f);
  const float e = __expf(2.0f * x);        // v_exp_f32
  return (e - 1.0f) * fast_rcp(e + 1.0f);
}

// ---------------------------------------------------------------------------
// Prep 1: pack weights f32 -> f16 (row-major, so B fragments are contiguous).
// ---------------------------------------------------------------------------
__global__ __launch_bounds__(256) void pack_weights_kernel(
    const float* __restrict__ pw, const float* __restrict__ w0,
    const float* __restrict__ w1, _Float16* __restrict__ pwh,
    _Float16* __restrict__ w0h, _Float16* __restrict__ w1h) {
  const int i = blockIdx.x * 256 + threadIdx.x;
  if (i < MLP_ * FD_)  pwh[i] = (_Float16)pw[i];
  if (i < HID_ * MLP_) w0h[i] = (_Float16)w0[i];
  if (i < MLP_ * HID_) w1h[i] = (_Float16)w1[i];
}

// ---------------------------------------------------------------------------
// Prep 2: softmax(agg) -> wts (d_ws), and zero d_out. Single block.
// ---------------------------------------------------------------------------
__global__ __launch_bounds__(256) void softmax_zero_kernel(
    const float* __restrict__ agg, float* __restrict__ wts,
    float* __restrict__ out) {
  __shared__ float red[256];
  const int t = threadIdx.x;

  float m = -1e30f;
  for (int i = t; i < NW_; i += 256) m = fmaxf(m, agg[i]);
  red[t] = m; __syncthreads();
  for (int s = 128; s > 0; s >>= 1) {
    if (t < s) red[t] = fmaxf(red[t], red[t + s]);
    __syncthreads();
  }
  const float mx = red[0];
  __syncthreads();

  float sum = 0.0f;
  for (int i = t; i < NW_; i += 256) sum += __expf(agg[i] - mx);
  red[t] = sum; __syncthreads();
  for (int s = 128; s > 0; s >>= 1) {
    if (t < s) red[t] += red[t + s];
    __syncthreads();
  }
  const float inv = 1.0f / red[0];

  for (int i = t; i < NW_; i += 256) wts[i] = __expf(agg[i] - mx) * inv;
  if (t < NB_) out[t] = 0.0f;
}

// ---------------------------------------------------------------------------
// Fused kernel: one wave per (batch, 16-row w tile). 2 waves / block.
// ---------------------------------------------------------------------------
__global__ __launch_bounds__(64) void snake_tcn_fused(
    const float* __restrict__ x,   const _Float16* __restrict__ pwh,
    const float* __restrict__ pb,  const _Float16* __restrict__ w0h,
    const float* __restrict__ b0,  const float* __restrict__ a0,
    const _Float16* __restrict__ w1h, const float* __restrict__ b1,
    const float* __restrict__ a1,  const float* __restrict__ wo,
    const float* __restrict__ bo,  const float* __restrict__ wts,
    float* __restrict__ out) {
  // Wave-private LDS (16B aligned for ds_load_b128 fragment reads):
  __shared__ alignas(16) _Float16 sfeat[2][16 * FD_];        // 32 KB (feat / h0)
  __shared__ alignas(16) _Float16 sp[2][16 * HID_];          //  4 KB (proj act)
  __shared__ alignas(16) _Float16 sxw[2][NC_ * WINLEN_];     // 2.5 KB (x windows)

  const int wave  = threadIdx.x >> 5;
  const int lane  = threadIdx.x & 31;
  const int hi    = lane >> 4;
  const int ln    = lane & 15;
  const int tile  = blockIdx.x * 2 + wave;
  if (tile >= NTILES_) return;              // whole wave exits (EXEC stays all-1)
  const int b     = blockIdx.y;
  const int wbase = tile * 16;

  _Float16* feat = sfeat[wave];
  _Float16* pbuf = sp[wave];
  _Float16* swin = sxw[wave];

  // ---- Stage x windows into LDS: 8 channels x 80 samples, f16 -------------
  // Lanes 0..19 each fetch one aligned float4 per channel.
  if (lane < 20) {
    const int base = wbase + lane * 4;
    for (int c = 0; c < NC_; ++c) {
      const float* xc = x + ((size_t)b * NC_ + c) * NT_;
      float4 v;
      if (base + 3 < NT_) {
        v = *(const float4*)(xc + base);    // 16B-aligned (wbase,lane*4 mult of 4)
      } else {                              // tail tile: clamped scalar loads
        int i0 = base + 0; if (i0 > NT_ - 1) i0 = NT_ - 1;
        int i1 = base + 1; if (i1 > NT_ - 1) i1 = NT_ - 1;
        int i2 = base + 2; if (i2 > NT_ - 1) i2 = NT_ - 1;
        int i3 = base + 3; if (i3 > NT_ - 1) i3 = NT_ - 1;
        v.x = xc[i0]; v.y = xc[i1]; v.z = xc[i2]; v.w = xc[i3];
      }
      _Float16* d = swin + c * WINLEN_ + lane * 4;
      d[0] = (_Float16)v.x; d[1] = (_Float16)v.y;
      d[2] = (_Float16)v.z; d[3] = (_Float16)v.w;
    }
  }
  // Same-wave LDS ordering: stores precede loads in program order, no barrier.

  // ---- DFT basis fragments (B layout), independent of channel -------------
  // rfft: F[f] = sum_k x[k]*(cos(2pi f k/64) - i sin(2pi f k/64))
  v16h bre[2][2], bim[2][2];                // [ntile(f 0-15 / 16-31)][kchunk]
  {
    const float step = 6.28318530717958647692f / (float)NK_;
#pragma unroll
    for (int nt = 0; nt < 2; ++nt) {
      const float ff = (float)(nt * 16 + ln);
#pragma unroll
      for (int kc = 0; kc < 2; ++kc) {
#pragma unroll
        for (int i = 0; i < 16; ++i) {
          const int k = kc * 32 + hi * 16 + i;   // bmapK
          const float ang = step * ff * (float)k;
          bre[nt][kc][i] = (_Float16)__cosf(ang);
          bim[nt][kc][i] = (_Float16)(-__sinf(ang));
        }
      }
    }
  }

  // ---- Per-channel sliding-window DFT + feature build ---------------------
  for (int c = 0; c < NC_; ++c) {
    // A fragments from staged window: win[w=ln][k] = swin[c*80 + ln + k]
    v16h afrag[2];
#pragma unroll
    for (int kc = 0; kc < 2; ++kc) {
#pragma unroll
      for (int i = 0; i < 16; ++i)
        afrag[kc][i] = swin[c * WINLEN_ + ln + kc * 32 + hi * 16 + i];
    }
#pragma unroll
    for (int nt = 0; nt < 2; ++nt) {
      v8f cr = {};
      v8f ci = {};
#pragma unroll
      for (int kc = 0; kc < 2; ++kc) {
        cr = wmma16(afrag[kc], bre[nt][kc], cr);
        ci = wmma16(afrag[kc], bim[nt][kc], ci);
      }
      // mag/phase -> interleaved feature tile in LDS (row-major 16x512)
      const int fcol = nt * 16 + ln;
#pragma unroll
      for (int j = 0; j < 8; ++j) {
        const float re  = cr[j];
        const float im  = ci[j];
        const float mag = 0.69314718056f *
                          __logf(1.0f + fast_sqrt(re * re + im * im)) *
                          1.44269504089f;  // ln via v_log: log2->ln fold
        const float ph  = atan2_over_pi(im, re);
        union { _Float16 h[2]; unsigned u; } pk2;
        pk2.h[0] = (_Float16)mag;
        pk2.h[1] = (_Float16)ph;
        const int wrow = j + hi * 8;
        *(unsigned*)(feat + wrow * FD_ + c * (2 * NF_) + 2 * fcol) = pk2.u;
      }
    }
  }

  // ---- proj: feat(16x512) @ pw^T(512x32) -> tanh * pi ---------------------
#pragma unroll
  for (int nt = 0; nt < 2; ++nt) {
    v8f acc = {};
    const int n = nt * 16 + ln;
#pragma unroll
    for (int kc = 0; kc < FD_ / 32; ++kc) {
      const v16h af = ldsA(feat + ln * FD_ + kc * 32, hi);
      const v16h bf = ldgBh(pwh + n * FD_ + kc * 32 + hi * 16);
      acc = wmma16(af, bf, acc);
    }
    const float bias = pb[n];
#pragma unroll
    for (int j = 0; j < 8; ++j) {
      const float v = fast_tanh(acc[j] + bias) * 3.14159265358979323846f;
      pbuf[(j + hi * 8) * HID_ + n] = (_Float16)v;
    }
  }

  // ---- h0 = snake(p(16x32) @ w0^T(32x64) + b0, a0), store into feat LDS ---
#pragma unroll
  for (int nt = 0; nt < 4; ++nt) {
    v8f acc = {};
    const int n = nt * 16 + ln;
    const v16h af = ldsA(pbuf + ln * HID_, hi);
    const v16h bf = ldgBh(w0h + n * MLP_ + hi * 16);
    acc = wmma16(af, bf, acc);
    const float bias = b0[n];
    const float aa   = fabsf(a0[n]) + 1e-8f;
    const float raa  = fast_rcp(aa);
#pragma unroll
    for (int j = 0; j < 8; ++j) {
      float v = acc[j] + bias;
      const float s = __sinf(aa * v);
      v += s * s * raa;                     // snake activation
      feat[(j + hi * 8) * FD_ + n] = (_Float16)v;   // 16x64 tile, stride FD_
    }
  }

  // ---- h1 = snake(h0(16x64) @ w1^T(64x32) + b1, a1) -----------------------
  v8f h1acc[2];
#pragma unroll
  for (int nt = 0; nt < 2; ++nt) {
    v8f acc = {};
    const int n = nt * 16 + ln;
#pragma unroll
    for (int kc = 0; kc < 2; ++kc) {
      const v16h af = ldsA(feat + ln * FD_ + kc * 32, hi);
      const v16h bf = ldgBh(w1h + n * HID_ + kc * 32 + hi * 16);
      acc = wmma16(af, bf, acc);
    }
    const float bias = b1[n];
    const float aa   = fabsf(a1[n]) + 1e-8f;
    const float raa  = fast_rcp(aa);
#pragma unroll
    for (int j = 0; j < 8; ++j) {
      float v = acc[j] + bias;
      const float s = __sinf(aa * v);
      h1acc[nt][j] = v + s * s * raa;
    }
  }

  // ---- out_t = h1 @ wo^T + bo; weighted sum over w -> out[b] --------------
  const float wn0 = wo[ln];
  const float wn1 = wo[16 + ln];
  float part[8];
#pragma unroll
  for (int j = 0; j < 8; ++j)
    part[j] = h1acc[0][j] * wn0 + h1acc[1][j] * wn1;
  // butterfly sum over the 16 columns held inside each lane-half
#pragma unroll
  for (int off = 1; off < 16; off <<= 1) {
#pragma unroll
    for (int j = 0; j < 8; ++j)
      part[j] += __shfl_xor(part[j], off, 32);
  }
  if (ln == 0) {                            // lanes 0 (rows 0-7) and 16 (rows 8-15)
    const float bias_o = bo[0];
    float s = 0.0f;
#pragma unroll
    for (int j = 0; j < 8; ++j) {
      const int w = wbase + j + hi * 8;
      if (w < NW_) s += (part[j] + bias_o) * wts[w];
    }
    atomicAdd(&out[b], s);
  }
}

// ---------------------------------------------------------------------------
extern "C" void kernel_launch(void* const* d_in, const int* in_sizes, int n_in,
                              void* d_out, int out_size, void* d_ws, size_t ws_size,
                              hipStream_t stream) {
  const float* x    = (const float*)d_in[0];
  const float* pw   = (const float*)d_in[1];
  const float* pb   = (const float*)d_in[2];
  const float* w0   = (const float*)d_in[3];
  const float* b0   = (const float*)d_in[4];
  const float* a0   = (const float*)d_in[5];
  const float* w1   = (const float*)d_in[6];
  const float* b1   = (const float*)d_in[7];
  const float* a1   = (const float*)d_in[8];
  const float* wo   = (const float*)d_in[9];
  const float* bo   = (const float*)d_in[10];
  const float* agg  = (const float*)d_in[11];
  float* out = (float*)d_out;

  float*     wts = (float*)d_ws;                               // 8 KB
  _Float16*  pwh = (_Float16*)((char*)d_ws + WS_PWH_OFF_);     // 32 KB
  _Float16*  w0h = (_Float16*)((char*)d_ws + WS_W0H_OFF_);     //  4 KB
  _Float16*  w1h = (_Float16*)((char*)d_ws + WS_W1H_OFF_);     //  4 KB

  pack_weights_kernel<<<(MLP_ * FD_ + 255) / 256, 256, 0, stream>>>(
      pw, w0, w1, pwh, w0h, w1h);
  softmax_zero_kernel<<<1, 256, 0, stream>>>(agg, wts, out);

  dim3 grid((NTILES_ + 1) / 2, NB_);       // 2 waves (tiles) per 64-thread block
  snake_tcn_fused<<<grid, 64, 0, stream>>>(x, pwh, pb, w0h, b0, a0,
                                           w1h, b1, a1, wo, bo, wts, out);
}
